// GCN_Encoder_31576599560347
// MI455X (gfx1250) — matrix operand order repaired and checked by reading, not compiled, not run on hardware
//
#include <hip/hip_runtime.h>
#include <hip/hip_bf16.h>

// ---------- raw vector types (trivial, union-safe) ----------
typedef __attribute__((ext_vector_type(16))) __bf16    v16bf;
typedef __attribute__((ext_vector_type(8)))  __bf16    bf16x8;
typedef __attribute__((ext_vector_type(2)))  __bf16    bf16x2;
typedef __attribute__((ext_vector_type(8)))  float     v8f;
typedef __attribute__((ext_vector_type(8)))  float     f32x8;
typedef __attribute__((ext_vector_type(4)))  float     f32x4;
typedef __attribute__((ext_vector_type(2)))  float     f32x2;
typedef __attribute__((ext_vector_type(4)))  unsigned  u32x4;

// ---------- bf16 helpers (vector conversions -> v_cvt_pk_bf16_f32) ----------
__device__ __forceinline__ unsigned short f2bf(float f) {
    __bf16 b = (__bf16)f;
    return __builtin_bit_cast(unsigned short, b);
}
__device__ __forceinline__ u32x4 cvt8bf(f32x8 v) {
    bf16x8 b = __builtin_convertvector(v, bf16x8);
    return __builtin_bit_cast(u32x4, b);
}

// ---------- problem constants ----------
#define GRIDX   28
#define NPATCH  784          // 28*28  (multiple of 16)
#define BATCH   64
#define MROWS   50176        // 64*784
#define HID     256
#define KEMB    192
#define IMG     224
#define IMGSZ   (224*224)

// ============================================================
// 1) Weight shuffle: W[K,256] f32 -> fragment-major bf16.
//    Fragment (kcIdx, ntile): 32 lanes x 16 elems contiguous (1KB).
//    lane: col = ntile*16 + (lane&15); k = kcIdx*32 + (e&7) + 8*(lane>>4) + 16*(e>>3)
// ============================================================
__global__ __launch_bounds__(256) void shuffle_w_kernel(
    const float* __restrict__ W, unsigned short* __restrict__ dst, int total)
{
    int idx = blockIdx.x * 256 + threadIdx.x;     // [0, K*256)
    if (idx >= total) return;
    int e     = idx & 15;
    int lane  = (idx >> 4) & 31;
    int fid   = idx >> 9;          // kcIdx*16 + ntile
    int nt    = fid & 15;
    int kcIdx = fid >> 4;
    int k = kcIdx * 32 + (e & 7) + 8 * (lane >> 4) + 16 * (e >> 3);
    int n = nt * 16 + (lane & 15);
    dst[idx] = f2bf(W[(size_t)k * HID + n]);
}

// ============================================================
// shared epilogue: lane owns col = waveN + t*16 + l16, rows r + 8*hh
// MODE 0: -> bf16 ; MODE 1: +bias+posenc -> bf16 ; MODE 2: +bias -> f32
// A 16-aligned row tile never crosses a batch boundary (784 % 16 == 0),
// so node = node0 + r inside the tile.
// ============================================================
template<int MODE>
__device__ __forceinline__ void store_tile(
    v8f* acc, int waveM, int waveN, int l16, int hh,
    const float* __restrict__ bias, void* __restrict__ Cout)
{
    const int row0  = waveM + 8 * hh;
    const int node0 = row0 % NPATCH;          // one modulo per tile
    #pragma unroll
    for (int t = 0; t < 4; ++t) {
        int col = waveN + t * 16 + l16;
        float bv  = (MODE != 0) ? bias[col] : 0.0f;
        float div = 0.0f;
        if (MODE == 1) {
            int dpair = col & ~1;
            div = __expf((float)dpair * (-9.210340371976184f / 256.0f));
        }
        #pragma unroll
        for (int r = 0; r < 8; ++r) {
            int row = row0 + r;
            float v = acc[t][r];
            if (MODE == 1) {
                float ang = (float)(node0 + r) * div;
                v += bv + ((col & 1) ? __cosf(ang) : __sinf(ang));
                ((unsigned short*)Cout)[(size_t)row * HID + col] = f2bf(v);
            } else if (MODE == 2) {
                ((float*)Cout)[(size_t)row * HID + col] = v + bv;
            } else {
                ((unsigned short*)Cout)[(size_t)row * HID + col] = f2bf(v);
            }
        }
    }
}

// ============================================================
// 2) Embed GEMM, A gathered directly from x (no patchify pass).
//    C[row, :] = patchrow(x) @ W_embed + b_embed + posenc -> bf16 h
//    K = 192.  Each A k-group of 8 = 8 consecutive pixels of a patch row;
//    the 8*hh k-shift is exactly one image row (folds into base pointer).
// ============================================================
__global__ __launch_bounds__(256) void gemm_embed_kernel(
    const float* __restrict__ x,
    const unsigned short* __restrict__ Bshuf,
    const float* __restrict__ bias,
    unsigned short* __restrict__ Cout)
{
    const int tid  = threadIdx.x;
    const int wave = tid >> 5;
    const int lane = tid & 31;
    const int hh   = lane >> 4;
    const int l16  = lane & 15;

    const int waveM = blockIdx.x * 64 + (wave >> 1) * 16;
    const int waveN = blockIdx.y * 128 + (wave & 1) * 64;
    const int ntBase = (blockIdx.y * 8) + (wave & 1) * 4;

    // this lane's A row -> (b, i, j) -> base pixel pointer (+hh row shift)
    int row  = waveM + l16;
    int b    = row / NPATCH;
    int node = row % NPATCH;
    int pi = node / GRIDX, pj = node % GRIDX;
    const float* xh = x + (size_t)b * 3 * IMGSZ + (size_t)(pi * 8 + hh) * IMG + pj * 8;

    // per-lane B base; fragment (kcIdx, t4) at constant byte offset
    const char* bb = (const char*)Bshuf + (size_t)ntBase * 1024 + (size_t)lane * 32;

    v8f acc[4] = {};

    #pragma unroll
    for (int kcIdx = 0; kcIdx < KEMB / 32; ++kcIdx) {
        union { u32x4 q[2]; v16bf v; } a;
        #pragma unroll
        for (int g = 0; g < 2; ++g) {
            const int kb0 = kcIdx * 32 + 16 * g;            // constant
            const int c   = kb0 >> 6;
            const int pp0 = (kb0 & 63) >> 3;
            const float* src = xh + c * IMGSZ + pp0 * IMG;  // imm offset
            union { f32x4 v4[2]; f32x8 v8; } t;
            t.v4[0] = *(const f32x4*)(src);
            t.v4[1] = *(const f32x4*)(src + 4);
            a.q[g] = cvt8bf(t.v8);                          // 4x v_cvt_pk_bf16_f32
        }
        #pragma unroll
        for (int t4 = 0; t4 < 4; ++t4) {
            const u32x4* p = (const u32x4*)(bb + kcIdx * 16384 + t4 * 1024);
            union { u32x4 q[2]; v16bf v; } bfr;
            bfr.q[0] = p[0];
            bfr.q[1] = p[1];
            acc[t4] = __builtin_amdgcn_wmma_f32_16x16x32_bf16(
                false, a.v, false, bfr.v, (short)0, acc[t4], false, false);
        }
    }
    store_tile<1>(acc, waveM, waveN, l16, hh, bias, Cout);
}

// ============================================================
// 3) Generic GEMM: C[M,256] = A[M,K](bf16 row-major) @ Bshuf(K,256)
//    MODE 0: -> bf16 hw ; MODE 2: +bias -> f32 out
//    All stream offsets are compile-time immediates off 2 base pointers.
// ============================================================
template<int K, int MODE>
__global__ __launch_bounds__(256) void gemm_kernel(
    const unsigned short* __restrict__ A,
    const unsigned short* __restrict__ Bshuf,
    const float* __restrict__ bias,
    void* __restrict__ Cout)
{
    const int tid  = threadIdx.x;
    const int wave = tid >> 5;
    const int lane = tid & 31;
    const int hh   = lane >> 4;
    const int l16  = lane & 15;

    const int waveM = blockIdx.x * 64 + (wave >> 1) * 16;
    const int waveN = blockIdx.y * 128 + (wave & 1) * 64;
    const int ntBase = (blockIdx.y * 8) + (wave & 1) * 4;

    v8f acc[4] = {};
    const char* ab = (const char*)(A + (size_t)(waveM + l16) * K + 8 * hh);
    const char* bb = (const char*)Bshuf + (size_t)ntBase * 1024 + (size_t)lane * 32;

    #pragma unroll
    for (int kcIdx = 0; kcIdx < K / 32; ++kcIdx) {
        // A fragment: lane (m=l16) holds K = kc + {0..7,16..23} + 8*hh
        union { u32x4 q[2]; v16bf v; } a;
        a.q[0] = *(const u32x4*)(ab + kcIdx * 64);
        a.q[1] = *(const u32x4*)(ab + kcIdx * 64 + 32);

        #pragma unroll
        for (int t4 = 0; t4 < 4; ++t4) {
            const u32x4* p = (const u32x4*)(bb + kcIdx * 16384 + t4 * 1024);
            union { u32x4 q[2]; v16bf v; } bfr;
            bfr.q[0] = p[0];
            bfr.q[1] = p[1];
            acc[t4] = __builtin_amdgcn_wmma_f32_16x16x32_bf16(
                false, a.v, false, bfr.v, (short)0, acc[t4], false, false);
        }
    }
    store_tile<MODE>(acc, waveM, waveN, l16, hh, bias, Cout);
}

// ============================================================
// 4) GCN aggregation as deterministic gather (no atomics), vectorized:
//    8 bf16 columns per thread, 8 rows per block; row is wave-uniform.
//    h[n] = relu( hw[n]/deg(n) + sum_nbr hw[nbr]*dinv(n)*dinv(nbr) + bias )
// ============================================================
__device__ __forceinline__ float node_deg(int i, int j) {
    return 1.0f + (j > 0) + (j < GRIDX - 1) + (i > 0) + (i < GRIDX - 1);
}

__device__ __forceinline__ void acc_row8(
    const unsigned short* __restrict__ p, float w, float* acc)
{
    u32x4 d = *(const u32x4*)p;
    #pragma unroll
    for (int q = 0; q < 4; ++q) {
        acc[2 * q]     += __uint_as_float(d[q] << 16) * w;
        acc[2 * q + 1] += __uint_as_float(d[q] & 0xffff0000u) * w;
    }
}

__global__ __launch_bounds__(256) void aggregate_kernel(
    const unsigned short* __restrict__ hw,
    const float* __restrict__ bias,
    unsigned short* __restrict__ hout)
{
    int tid  = threadIdx.x;
    int row  = blockIdx.x * 8 + (tid >> 5);   // wave-uniform
    int colg = (tid & 31) * 8;
    int b    = row / NPATCH;
    int node = row % NPATCH;
    int i = node / GRIDX, j = node % GRIDX;

    float deg  = node_deg(i, j);
    float dinv = rsqrtf(deg);

    float acc[8] = {};
    acc_row8(hw + (size_t)row * HID + colg, 1.0f / deg, acc);

    const unsigned short* base = hw + (size_t)b * NPATCH * HID + colg;
    if (j > 0)
        acc_row8(base + (size_t)(node - 1) * HID,     dinv * rsqrtf(node_deg(i, j - 1)), acc);
    if (j < GRIDX - 1)
        acc_row8(base + (size_t)(node + 1) * HID,     dinv * rsqrtf(node_deg(i, j + 1)), acc);
    if (i > 0)
        acc_row8(base + (size_t)(node - GRIDX) * HID, dinv * rsqrtf(node_deg(i - 1, j)), acc);
    if (i < GRIDX - 1)
        acc_row8(base + (size_t)(node + GRIDX) * HID, dinv * rsqrtf(node_deg(i + 1, j)), acc);

    f32x4 b0 = *(const f32x4*)(bias + colg);
    f32x4 b1 = *(const f32x4*)(bias + colg + 4);

    f32x8 r;
    #pragma unroll
    for (int q = 0; q < 4; ++q) {
        r[q]     = fmaxf(acc[q]     + b0[q], 0.0f);
        r[q + 4] = fmaxf(acc[q + 4] + b1[q], 0.0f);
    }
    *(u32x4*)(hout + (size_t)row * HID + colg) = cvt8bf(r);
}

// ============================================================
// host launcher
// ============================================================
extern "C" void kernel_launch(void* const* d_in, const int* in_sizes, int n_in,
                              void* d_out, int out_size, void* d_ws, size_t ws_size,
                              hipStream_t stream)
{
    const float* x       = (const float*)d_in[0];
    const float* W_embed = (const float*)d_in[1];
    const float* b_embed = (const float*)d_in[2];
    const float* W_gcn   = (const float*)d_in[3];   // [3,256,256]
    const float* b_gcn   = (const float*)d_in[4];   // [3,256]
    const float* W_out   = (const float*)d_in[5];
    const float* b_out   = (const float*)d_in[6];
    (void)in_sizes; (void)n_in; (void)out_size; (void)ws_size;

    // workspace layout (bytes, all 16B aligned)
    char* ws = (char*)d_ws;
    unsigned short* h   = (unsigned short*)(ws);                                   // 50176*256*2
    unsigned short* hw  = (unsigned short*)((char*)h  + (size_t)MROWS * HID * 2);  // 50176*256*2
    unsigned short* We  = (unsigned short*)((char*)hw + (size_t)MROWS * HID * 2);  // 192*256*2
    unsigned short* Wg  = (unsigned short*)((char*)We + (size_t)KEMB * HID * 2);   // 3*256*256*2
    unsigned short* Wo  = (unsigned short*)((char*)Wg + (size_t)3 * HID * HID * 2);// 256*256*2

    // 1) shuffle weights into WMMA B-fragment layout (bf16)
    shuffle_w_kernel<<<(KEMB * HID + 255) / 256, 256, 0, stream>>>(W_embed, We, KEMB * HID);
    for (int l = 0; l < 3; ++l)
        shuffle_w_kernel<<<(HID * HID + 255) / 256, 256, 0, stream>>>(
            W_gcn + (size_t)l * HID * HID, Wg + (size_t)l * HID * HID, HID * HID);
    shuffle_w_kernel<<<(HID * HID + 255) / 256, 256, 0, stream>>>(W_out, Wo, HID * HID);

    dim3 gg(MROWS / 64, HID / 128);   // (784, 2)

    // 2) fused patchify + embed + positional encoding -> h (bf16)
    gemm_embed_kernel<<<gg, 256, 0, stream>>>(x, We, b_embed, h);

    // 3) three GCN layers
    for (int l = 0; l < 3; ++l) {
        gemm_kernel<HID, 0><<<gg, 256, 0, stream>>>(h, Wg + (size_t)l * HID * HID, nullptr, hw);
        aggregate_kernel<<<MROWS / 8, 256, 0, stream>>>(hw, b_gcn + l * HID, h);
    }

    // 4) output projection -> f32
    gemm_kernel<HID, 2><<<gg, 256, 0, stream>>>(h, Wo, b_out, (float*)d_out);
}
